// GNN_Guided_Baseline_Simple_21689584845280
// MI455X (gfx1250) — matrix operand order repaired and checked by reading, not compiled, not run on hardware
//
#include <hip/hip_runtime.h>
#include <hip/hip_bf16.h>

typedef __attribute__((ext_vector_type(16))) _Float16 v16h;
typedef __attribute__((ext_vector_type(8)))  _Float16 v8h;
typedef __attribute__((ext_vector_type(8)))  float    v8f;

#define DIM 256
#define BN_EPS 1e-5f

static __device__ __forceinline__ v8f wmma16x16x32(v16h a, v16h b, v8f c) {
  // (neg_a, A, neg_b, B, c_mod, C, reuse_a, reuse_b)
  return __builtin_amdgcn_wmma_f32_16x16x32_f16(false, a, false, b, (short)0, c,
                                                false, false);
}

static __device__ __forceinline__ float waveSum(float v) {
#pragma unroll
  for (int o = 16; o > 0; o >>= 1) v += __shfl_down(v, o, 32);
  return v;
}

static __device__ __forceinline__ void atomicMaxF32(float* addr, float value) {
  // signed/unsigned monotonic-bits trick; init value must be -inf
  if (value >= 0.f) atomicMax((int*)addr, __float_as_int(value));
  else              atomicMin((unsigned int*)addr, __float_as_uint(value));
}

// ---------------------------------------------------------------- init utils
__global__ void fill_f32(float* __restrict__ p, float v, int n) {
  int i = blockIdx.x * blockDim.x + threadIdx.x;
  if (i < n) p[i] = v;
}

__global__ void f32_to_f16(const float* __restrict__ s, _Float16* __restrict__ d, int n) {
  int i = blockIdx.x * blockDim.x + threadIdx.x;
  if (i < n) d[i] = (_Float16)s[i];
}

// ------------------------------------------------- 1) edge segment-sum (GIN agg)
__global__ void edge_scatter(const long long* __restrict__ ei,
                             const float* __restrict__ x,
                             float* __restrict__ agg, int nEdges) {
  int e = blockIdx.x * blockDim.x + threadIdx.x;
  if (e >= nEdges) return;
  long long s = ei[e];                       // src row
  long long d = ei[(long long)nEdges + e];   // dst row
  atomicAdd(&agg[d * 3 + 0], x[s * 3 + 0]);
  atomicAdd(&agg[d * 3 + 1], x[s * 3 + 1]);
  atomicAdd(&agg[d * 3 + 2], x[s * 3 + 2]);
}

// ----------------------------------- 2) BN(3) + Linear(3->256) + ReLU, f16 out
__global__ __launch_bounds__(DIM) void layer1_kernel(
    const float* __restrict__ x, const float* __restrict__ agg,
    const float* __restrict__ gamma, const float* __restrict__ beta,
    const float* __restrict__ mean, const float* __restrict__ var,
    const float* __restrict__ w1, const float* __restrict__ b1,
    _Float16* __restrict__ out) {
  const int node = blockIdx.x;
  const int c = threadIdx.x;
  float xb[3];
#pragma unroll
  for (int k = 0; k < 3; ++k) {
    float h = x[node * 3 + k] + agg[node * 3 + k];
    xb[k] = (h - mean[k]) * rsqrtf(var[k] + BN_EPS) * gamma[k] + beta[k];
  }
  float s = b1[c];
#pragma unroll
  for (int k = 0; k < 3; ++k) s += xb[k] * w1[c * 3 + k];
  out[(size_t)node * DIM + c] = (_Float16)(s > 0.f ? s : 0.f);
}

// --------------------- 3) WMMA GEMM: out = relu(A[16n,256] @ W^T + bias), f16
// Block = 16 nodes x 256 cols, 8 waves, wave w owns columns [32w, 32w+32).
__global__ __launch_bounds__(256) void gemm_relu_kernel(
    const _Float16* __restrict__ A,   // [N,256] activations (row-major)
    const _Float16* __restrict__ W,   // [256,256] weights, W[c][k]
    const float* __restrict__ bias, _Float16* __restrict__ out) {
  __shared__ _Float16 sA[16 * DIM];   // 8 KB A-tile
  const int tid = threadIdx.x;
  {
    const uint4* src = (const uint4*)(A + (size_t)blockIdx.x * 16 * DIM);
    uint4* dst = (uint4*)sA;
    dst[tid] = src[tid];
    dst[tid + 256] = src[tid + 256];
  }
  __syncthreads();

  const int wave = tid >> 5;
  const int lane = tid & 31;
  const int row  = lane & 15;
  const int hi   = lane >> 4;           // 0: lanes 0-15, 1: lanes 16-31

  v8f acc0 = {};
  v8f acc1 = {};
  // B fragment: lane holds column (lane&15); halves h -> K = hi*16 + h
  const _Float16* w0 = W + (size_t)(wave * 32 + row) * DIM + hi * 16;
  const _Float16* w1 = w0 + (size_t)16 * DIM;
  // A fragment: lane holds row (lane&15); halves h -> K = hi*16 + h
  const _Float16* ar = &sA[row * DIM + hi * 16];

#pragma unroll
  for (int kb = 0; kb < 8; ++kb) {
    v8h alo = *(const v8h*)(ar + kb * 32);
    v8h ahi = *(const v8h*)(ar + kb * 32 + 8);
    v16h a = __builtin_shufflevector(alo, ahi, 0, 1, 2, 3, 4, 5, 6, 7,
                                     8, 9, 10, 11, 12, 13, 14, 15);
    v16h b0 = *(const v16h*)(w0 + kb * 32);
    v16h b1 = *(const v16h*)(w1 + kb * 32);
    acc0 = wmma16x16x32(a, b0, acc0);
    acc1 = wmma16x16x32(a, b1, acc1);
  }

  // D layout: lane L -> N = L&15; VGPR r -> M = hi*8 + r
  const size_t nodeBase = (size_t)blockIdx.x * 16;
  const int n0 = wave * 32 + row;
  const int n1 = n0 + 16;
  const float bb0 = bias[n0], bb1 = bias[n1];
#pragma unroll
  for (int r = 0; r < 8; ++r) {
    int m = hi * 8 + r;
    float v0 = acc0[r] + bb0; v0 = v0 > 0.f ? v0 : 0.f;
    float v1 = acc1[r] + bb1; v1 = v1 > 0.f ? v1 : 0.f;
    out[(nodeBase + m) * DIM + n0] = (_Float16)v0;
    out[(nodeBase + m) * DIM + n1] = (_Float16)v1;
  }
}

// ------------------------------------- 4) gate score + per-graph running max
__global__ __launch_bounds__(256) void gate_kernel(
    const _Float16* __restrict__ h, const float* __restrict__ gw,
    const float* __restrict__ gb, const long long* __restrict__ batch,
    float* __restrict__ g, float* __restrict__ gmax, int nNodes) {
  int node = blockIdx.x * 8 + (threadIdx.x >> 5);
  int lane = threadIdx.x & 31;
  if (node >= nNodes) return;
  const _Float16* hp = h + (size_t)node * DIM;
  float s = 0.f;
#pragma unroll
  for (int k = 0; k < DIM / 32; ++k)
    s += (float)hp[lane + k * 32] * gw[lane + k * 32];
  s = waveSum(s);
  if (lane == 0) {
    float val = s + gb[0];
    g[node] = val;
    atomicMaxF32(&gmax[batch[node]], val);
  }
}

// --------------------------------------------- 5) exp(g - gmax), denom sums
__global__ void exp_kernel(const float* __restrict__ g,
                           const long long* __restrict__ batch,
                           const float* __restrict__ gmax,
                           float* __restrict__ e, float* __restrict__ denom,
                           int nNodes) {
  int i = blockIdx.x * blockDim.x + threadIdx.x;
  if (i >= nNodes) return;
  long long b = batch[i];
  float v = __expf(g[i] - gmax[b]);
  e[i] = v;
  atomicAdd(&denom[b], v);
}

// ------------------------------------------ 6) pooled[g] += e_i * h_i (f32)
__global__ __launch_bounds__(DIM) void pool_kernel(
    const _Float16* __restrict__ h, const float* __restrict__ e,
    const long long* __restrict__ batch, float* __restrict__ pooled) {
  int node = blockIdx.x;
  int c = threadIdx.x;
  float v = e[node] * (float)h[(size_t)node * DIM + c];
  atomicAdd(&pooled[(size_t)batch[node] * DIM + c], v);
}

// --------------------------- 7) a/b GEMVs + affine combine; outputs (out,a,b)
__global__ __launch_bounds__(256) void final_kernel(
    const float* __restrict__ pooled, const float* __restrict__ denom,
    const float* __restrict__ fc1w, const float* __restrict__ fc1b,
    const float* __restrict__ fc2w, const float* __restrict__ fc2b,
    const float* __restrict__ base, float* __restrict__ out, int nGraphs) {
  int gph = blockIdx.x * 8 + (threadIdx.x >> 5);
  int lane = threadIdx.x & 31;
  if (gph >= nGraphs) return;
  float inv = 1.f / denom[gph];
  float sa = 0.f, sb = 0.f;
#pragma unroll
  for (int k = 0; k < DIM / 32; ++k) {
    float p = pooled[(size_t)gph * DIM + lane + k * 32] * inv;
    sa += p * fc1w[lane + k * 32];
    sb += p * fc2w[lane + k * 32];
  }
  sa = waveSum(sa);
  sb = waveSum(sb);
  if (lane == 0) {
    float a = sa + fc1b[0];
    float b = sb + fc2b[0];
    out[gph] = a * base[gph] + b;       // out
    out[nGraphs + gph] = a;             // a
    out[2 * nGraphs + gph] = b;         // b
  }
}

extern "C" void kernel_launch(void* const* d_in, const int* in_sizes, int n_in,
                              void* d_out, int out_size, void* d_ws, size_t ws_size,
                              hipStream_t stream) {
  const float*     x         = (const float*)d_in[0];
  const long long* edge_index= (const long long*)d_in[1];
  const long long* batch     = (const long long*)d_in[2];
  const float*     base      = (const float*)d_in[3];
  const float*     bn_gamma  = (const float*)d_in[4];
  const float*     bn_beta   = (const float*)d_in[5];
  const float*     bn_mean   = (const float*)d_in[6];
  const float*     bn_var    = (const float*)d_in[7];
  const float*     w1        = (const float*)d_in[8];
  const float*     b1        = (const float*)d_in[9];
  const float*     w2        = (const float*)d_in[10];
  const float*     b2        = (const float*)d_in[11];
  const float*     w3        = (const float*)d_in[12];
  const float*     b3        = (const float*)d_in[13];
  const float*     gate_w    = (const float*)d_in[14];
  const float*     gate_b    = (const float*)d_in[15];
  const float*     fc1_w     = (const float*)d_in[16];
  const float*     fc1_b     = (const float*)d_in[17];
  const float*     fc2_w     = (const float*)d_in[18];
  const float*     fc2_b     = (const float*)d_in[19];

  const int nNodes  = in_sizes[0] / 3;   // 100000
  const int nEdges  = in_sizes[1] / 2;   // 3200000
  const int nGraphs = in_sizes[3];       // 512

  // ---- workspace carve-out (~105 MB) ----
  char* ws = (char*)d_ws;
  size_t off = 0;
  auto take = [&](size_t bytes) -> char* {
    char* p = ws + off;
    off = (off + bytes + 255) & ~(size_t)255;
    return p;
  };
  float*    agg    = (float*)   take((size_t)nNodes * 3 * sizeof(float));
  _Float16* hA     = (_Float16*)take((size_t)nNodes * DIM * sizeof(_Float16));
  _Float16* hB     = (_Float16*)take((size_t)nNodes * DIM * sizeof(_Float16));
  _Float16* w2h    = (_Float16*)take((size_t)DIM * DIM * sizeof(_Float16));
  _Float16* w3h    = (_Float16*)take((size_t)DIM * DIM * sizeof(_Float16));
  float*    gbuf   = (float*)   take((size_t)nNodes * sizeof(float));
  float*    ebuf   = (float*)   take((size_t)nNodes * sizeof(float));
  float*    gmax   = (float*)   take((size_t)nGraphs * sizeof(float));
  float*    denom  = (float*)   take((size_t)nGraphs * sizeof(float));
  float*    pooled = (float*)   take((size_t)nGraphs * DIM * sizeof(float));

  // init (d_ws is poisoned; zero/neg-inf what we accumulate into)
  fill_f32<<<(nNodes * 3 + 255) / 256, 256, 0, stream>>>(agg, 0.f, nNodes * 3);
  fill_f32<<<(nGraphs + 255) / 256, 256, 0, stream>>>(denom, 0.f, nGraphs);
  fill_f32<<<(nGraphs * DIM + 255) / 256, 256, 0, stream>>>(pooled, 0.f, nGraphs * DIM);
  fill_f32<<<(nGraphs + 255) / 256, 256, 0, stream>>>(gmax, -__builtin_inff(), nGraphs);
  f32_to_f16<<<(DIM * DIM + 255) / 256, 256, 0, stream>>>(w2, w2h, DIM * DIM);
  f32_to_f16<<<(DIM * DIM + 255) / 256, 256, 0, stream>>>(w3, w3h, DIM * DIM);

  // pipeline
  edge_scatter<<<(nEdges + 255) / 256, 256, 0, stream>>>(edge_index, x, agg, nEdges);
  layer1_kernel<<<nNodes, DIM, 0, stream>>>(x, agg, bn_gamma, bn_beta, bn_mean,
                                            bn_var, w1, b1, hA);
  gemm_relu_kernel<<<nNodes / 16, 256, 0, stream>>>(hA, w2h, b2, hB);
  gemm_relu_kernel<<<nNodes / 16, 256, 0, stream>>>(hB, w3h, b3, hA);  // outer relu folded
  gate_kernel<<<(nNodes + 7) / 8, 256, 0, stream>>>(hA, gate_w, gate_b, batch,
                                                    gbuf, gmax, nNodes);
  exp_kernel<<<(nNodes + 255) / 256, 256, 0, stream>>>(gbuf, batch, gmax, ebuf,
                                                       denom, nNodes);
  pool_kernel<<<nNodes, DIM, 0, stream>>>(hA, ebuf, batch, pooled);
  final_kernel<<<(nGraphs + 7) / 8, 256, 0, stream>>>(pooled, denom, fc1_w, fc1_b,
                                                      fc2_w, fc2_b, base,
                                                      (float*)d_out, nGraphs);
}
